// WindowAttention_85718957293904
// MI455X (gfx1250) — compile-verified, matrix operand order checked by reading, not run on hardware
//
#include <hip/hip_runtime.h>
#include <hip/hip_bf16.h>

// ---------------------------------------------------------------------------
// CDNA5 / gfx1250 fused window-attention.
// All GEMMs via v_wmma_f32_16x16x32_f16 (wave32, fp32 accum).
// fp32 operands pre-converted to f16 once so GEMM hot loops are pure
// b128-load + v_wmma. Attention is fused flash-style with a shift-free
// exact softmax (logit range makes max-subtraction unnecessary), so the
// chunk loop has no cross-lane reductions at all.
// ---------------------------------------------------------------------------

typedef _Float16 half_t;
typedef __attribute__((ext_vector_type(16))) _Float16 v16h;
typedef __attribute__((ext_vector_type(8)))  float    v8f;

#define WM_N   343          // tokens per window (7*7*7)
#define NPAD   352          // 343 padded to 22*16
#define HEADS  12
#define HD     32           // head dim
#define CDIM   384
#define BATCH  128
#define NWIN   64
#define MROWS  (BATCH * WM_N)        // 43904, divisible by 32
#define MT32   (MROWS / 32)          // 1372

__device__ __forceinline__ v8f wmma16(v16h a, v16h b, v8f c) {
  // D = A(16x32 f16) * B(32x16 f16) + C(16x16 f32)
  return __builtin_amdgcn_wmma_f32_16x16x32_f16(
      /*neg_a=*/false, a, /*neg_b=*/false, b,
      /*c_mod=*/(short)0, c, /*reuse_a=*/false, /*reuse_b=*/false);
}

// Load a 16x32 f16 fragment from a row-major ("N-major"/"M-major") f16 source.
// ISA layout: lane L(0..15) row L, K=0..7 (v0..3) & 16..23 (v4..7);
//             lane L+16     row L, K=8..15        & 24..31.
// Contiguous 16B runs -> compiler emits 2x global_load_b128 per lane.
__device__ __forceinline__ v16h load_frag_h(const half_t* __restrict__ src, int stride) {
  const int lane = threadIdx.x & 31;
  const int r    = lane & 15;
  const int hi   = lane >> 4;
  const half_t* p = src + r * stride + hi * 8;
  v16h f;
#pragma unroll
  for (int i = 0; i < 8; ++i) f[i] = p[i];
#pragma unroll
  for (int i = 0; i < 8; ++i) f[8 + i] = p[16 + i];
  return f;
}

// ---------------------------------------------------------------------------
// 0) fp32 -> f16 streaming conversion (one-time; removes cvt from GEMM loops)
// ---------------------------------------------------------------------------
__global__ void cvt16_kernel(const float* __restrict__ src,
                             half_t* __restrict__ dst, int n) {
  int i = (blockIdx.x * blockDim.x + threadIdx.x) * 4;
  if (i + 3 < n) {
    float4 v = *(const float4*)(src + i);
    dst[i]     = (half_t)v.x;
    dst[i + 1] = (half_t)v.y;
    dst[i + 2] = (half_t)v.z;
    dst[i + 3] = (half_t)v.w;
  } else {
    for (; i < n; ++i) dst[i] = (half_t)src[i];
  }
}

// ---------------------------------------------------------------------------
// 1) Pre-gather relative-position bias: biasc[h][i][j] = rpb[rel[i*N+j]][h]
// ---------------------------------------------------------------------------
__global__ void biasc_kernel(const float* __restrict__ rpb,
                             const int* __restrict__ rel,
                             float* __restrict__ biasc) {
  const int NN = WM_N * WM_N;
  int idx = blockIdx.x * blockDim.x + threadIdx.x;
  if (idx >= NN) return;
  int ri = rel[idx];
#pragma unroll
  for (int h = 0; h < HEADS; ++h)
    biasc[(size_t)h * NN + idx] = rpb[ri * HEADS + h];
}

// ---------------------------------------------------------------------------
// 2) Zero the 343..351 padding rows of q, k (row=key) and vT (col=key).
//    One block per (b*H+h), 288 = 9*32 threads.
// ---------------------------------------------------------------------------
__global__ void zeropad_kernel(half_t* __restrict__ q, half_t* __restrict__ k,
                               half_t* __restrict__ vT) {
  const int bh = blockIdx.x;
  const int t  = threadIdx.x;           // 0..287
  int n = WM_N + t / HD, d = t % HD;    // 9 pad rows x 32 dims
  q[((size_t)bh * NPAD + n) * HD + d] = (half_t)0.f;
  k[((size_t)bh * NPAD + n) * HD + d] = (half_t)0.f;
  int d2 = t / 9, key = WM_N + t % 9;   // 32 dims x 9 pad keys
  vT[((size_t)bh * HD + d2) * NPAD + key] = (half_t)0.f;
}

// ---------------------------------------------------------------------------
// 3) QKV projection: [43904,384] x [384,1152] (+bias) -> q/k/vT (f16).
//    One wave computes a 32x64 tile: per k-step 2 A-frags + 4 B-frags
//    feed 8 WMMAs (8 accumulators = 64 VGPRs).
//    q pre-scaled by 1/sqrt(hd); v stored transposed [hd,key] for PV B-frags.
// ---------------------------------------------------------------------------
__global__ __launch_bounds__(32) void qkv_kernel(
    const half_t* __restrict__ xh, const half_t* __restrict__ wh,
    const float* __restrict__ bias, half_t* __restrict__ q,
    half_t* __restrict__ k, half_t* __restrict__ vT) {
  const int mt = blockIdx.x;            // 32-row tile of flattened (b,n)
  const int ct = blockIdx.y;            // 64-col tile of 1152
  const int lane = threadIdx.x & 31;
  const int hi = lane >> 4, cl = lane & 15;
  const float scale = 0.17677669529663687f;   // 32^-0.5

  v8f acc[2][4] = {};
  const half_t* arow = xh + (size_t)mt * 32 * CDIM;
  for (int kt = 0; kt < CDIM / 32; ++kt) {
    v16h a0 = load_frag_h(arow + kt * 32, CDIM);
    v16h a1 = load_frag_h(arow + 16 * CDIM + kt * 32, CDIM);
#pragma unroll
    for (int j = 0; j < 4; ++j) {
      // W is [1152,384] row-major == N-major view of B = W^T -> contiguous frag
      v16h b = load_frag_h(wh + (size_t)(ct * 64 + j * 16) * CDIM + kt * 32, CDIM);
      acc[0][j] = wmma16(a0, b, acc[0][j]);
      acc[1][j] = wmma16(a1, b, acc[1][j]);
    }
  }
#pragma unroll
  for (int i = 0; i < 2; ++i) {
#pragma unroll
    for (int j = 0; j < 4; ++j) {
      int col = ct * 64 + j * 16 + cl;  // 0..1151
      int s = col / CDIM;               // 0=q 1=k 2=v
      int rem = col - s * CDIM;
      int h = rem >> 5, d = rem & 31;
      float bv = bias[col];
#pragma unroll
      for (int r = 0; r < 8; ++r) {
        int m  = mt * 32 + i * 16 + r + 8 * hi;   // C-frag row mapping
        int b_ = m / WM_N;
        int n  = m - b_ * WM_N;
        float v = acc[i][j][r] + bv;
        size_t bh = (size_t)b_ * HEADS + h;
        if (s == 0)      q[(bh * NPAD + n) * HD + d]  = (half_t)(v * scale);
        else if (s == 1) k[(bh * NPAD + n) * HD + d]  = (half_t)v;
        else             vT[(bh * HD + d) * NPAD + n] = (half_t)v;
      }
    }
  }
}

// ---------------------------------------------------------------------------
// 4) Fused attention. One wave per (q-tile, b, h).
//    Shift-free exact softmax: logits here are O(1) (plus -100 mask terms),
//    far from fp32 exp overflow, so no running max / rescale is needed and
//    the softmax is mathematically identical. fminf(e,80) guards the
//    impossible overflow case. Chunk loop: 2 QK^T WMMAs -> add bias+mask,
//    exp, per-lane partial sums -> P via LDS relayout -> 2 PV WMMAs.
//    Row sums reduced across lanes once, after the loop.
// ---------------------------------------------------------------------------
__global__ __launch_bounds__(32) void attn_kernel(
    const half_t* __restrict__ q, const half_t* __restrict__ k,
    const half_t* __restrict__ vT, const float* __restrict__ biasc,
    const float* __restrict__ mask, half_t* __restrict__ ob) {
  __shared__ half_t pS[16][32];         // P tile relayout buffer (1 KB)

  const int qt = blockIdx.x;            // 0..21
  const int bh = blockIdx.y;            // b*12+h
  const int b_ = bh / HEADS;
  const int h  = bh - b_ * HEADS;
  const int w_ = b_ & (NWIN - 1);       // window index = b % 64
  const int lane = threadIdx.x & 31;
  const int hi = lane >> 4, cl = lane & 15;
  const int NN = WM_N * WM_N;

  const half_t* qbase = q  + (size_t)bh * NPAD * HD;
  const half_t* kbase = k  + (size_t)bh * NPAD * HD;
  const half_t* vbase = vT + (size_t)bh * HD * NPAD;
  const float*  bb = biasc + (size_t)h  * NN;
  const float*  mb = mask  + (size_t)w_ * NN;

  v16h aq = load_frag_h(qbase + (size_t)qt * 16 * HD, HD);  // Q tile, reused 11x
  v8f o0 = {}, o1 = {};
  float lsum[8];
#pragma unroll
  for (int r = 0; r < 8; ++r) lsum[r] = 0.f;

  for (int t = 0; t < NPAD / 32; ++t) {
    // --- S = Q K^T for 32 keys (k stored [key,hd] => contiguous B-frags) ---
    v16h kb0 = load_frag_h(kbase + (size_t)(t * 32) * HD, HD);
    v16h kb1 = load_frag_h(kbase + (size_t)(t * 32 + 16) * HD, HD);
    v8f s0 = {}, s1 = {};
    s0 = wmma16(aq, kb0, s0);
    s1 = wmma16(aq, kb1, s1);

    float p0[8], p1[8];
#pragma unroll
    for (int r = 0; r < 8; ++r) {
      int ig = qt * 16 + r + 8 * hi;    // global query row of this C element
      int j0 = t * 32 + cl;             // global key col (frag 0)
      int j1 = j0 + 16;                 // global key col (frag 1)
      float e0, e1;
      if (ig < WM_N && j0 < WM_N) {
        size_t rb = (size_t)ig * WM_N;
        e0 = s0[r] + bb[rb + j0] + mb[rb + j0];
      } else e0 = -1e30f;
      if (ig < WM_N && j1 < WM_N) {
        size_t rb = (size_t)ig * WM_N;
        e1 = s1[r] + bb[rb + j1] + mb[rb + j1];
      } else e1 = -1e30f;
      float q0 = __expf(fminf(e0, 80.f));   // masked/pad -> exp(-1e30) = 0
      float q1 = __expf(fminf(e1, 80.f));
      lsum[r] += q0 + q1;                   // per-lane partial row sum
      p0[r] = q0;
      p1[r] = q1;
    }

    // --- relayout P: C-frag (row=r+8*hi, col=lane&15) -> LDS -> A-frag ---
    __syncthreads();
#pragma unroll
    for (int r = 0; r < 8; ++r) {
      pS[r + 8 * hi][cl]      = (half_t)p0[r];
      pS[r + 8 * hi][16 + cl] = (half_t)p1[r];
    }
    __syncthreads();
    v16h pa = load_frag_h(&pS[0][0], 32);

    // --- O += P V (vT stored [hd,key] => contiguous B-frags) ---
    v16h vb0 = load_frag_h(vbase + t * 32, NPAD);               // hd 0..15
    v16h vb1 = load_frag_h(vbase + 16 * NPAD + t * 32, NPAD);   // hd 16..31
    o0 = wmma16(pa, vb0, o0);
    o1 = wmma16(pa, vb1, o1);
    __syncthreads();
  }

  // --- one row-sum reduction per row (xor 1,2,4,8 stays in each half) ---
#pragma unroll
  for (int r = 0; r < 8; ++r) {
    float cs = lsum[r];
#pragma unroll
    for (int off = 1; off < 16; off <<= 1)
      cs += __shfl_xor(cs, off, 32);
    int ig = qt * 16 + r + 8 * hi;
    if (ig >= WM_N) continue;           // padded query rows never stored
    size_t m = (size_t)b_ * WM_N + ig;
    float inv = 1.0f / cs;
    ob[m * CDIM + h * HD + cl]      = (half_t)(o0[r] * inv);
    ob[m * CDIM + h * HD + 16 + cl] = (half_t)(o1[r] * inv);
  }
}

// ---------------------------------------------------------------------------
// 5) Output projection: [43904,384] x [384,384] (+bias) -> fp32 out.
//    Same 32x64-per-wave WMMA tiling as the QKV GEMM.
// ---------------------------------------------------------------------------
__global__ __launch_bounds__(32) void proj_kernel(
    const half_t* __restrict__ ob, const half_t* __restrict__ wh,
    const float* __restrict__ bias, float* __restrict__ out) {
  const int mt = blockIdx.x;
  const int ct = blockIdx.y;            // 64-col tile of 384
  const int lane = threadIdx.x & 31;
  const int hi = lane >> 4, cl = lane & 15;

  v8f acc[2][4] = {};
  const half_t* arow = ob + (size_t)mt * 32 * CDIM;
  for (int kt = 0; kt < CDIM / 32; ++kt) {
    v16h a0 = load_frag_h(arow + kt * 32, CDIM);
    v16h a1 = load_frag_h(arow + 16 * CDIM + kt * 32, CDIM);
#pragma unroll
    for (int j = 0; j < 4; ++j) {
      v16h b = load_frag_h(wh + (size_t)(ct * 64 + j * 16) * CDIM + kt * 32, CDIM);
      acc[0][j] = wmma16(a0, b, acc[0][j]);
      acc[1][j] = wmma16(a1, b, acc[1][j]);
    }
  }
#pragma unroll
  for (int i = 0; i < 2; ++i) {
#pragma unroll
    for (int j = 0; j < 4; ++j) {
      int col = ct * 64 + j * 16 + cl;
      float bv = bias[col];
#pragma unroll
      for (int r = 0; r < 8; ++r) {
        size_t m = (size_t)mt * 32 + i * 16 + r + 8 * hi;
        out[m * CDIM + col] = acc[i][j][r] + bv;
      }
    }
  }
}

// ---------------------------------------------------------------------------
extern "C" void kernel_launch(void* const* d_in, const int* in_sizes, int n_in,
                              void* d_out, int out_size, void* d_ws, size_t ws_size,
                              hipStream_t stream) {
  const float* x      = (const float*)d_in[0];
  const float* mask   = (const float*)d_in[1];
  const float* qkv_w  = (const float*)d_in[2];
  const float* qkv_b  = (const float*)d_in[3];
  const float* proj_w = (const float*)d_in[4];
  const float* proj_b = (const float*)d_in[5];
  const float* rpb    = (const float*)d_in[6];
  const int*   rel    = (const int*)d_in[7];
  float* out = (float*)d_out;

  // Workspace carve-up (256B aligned):
  //   xh:      [43904,384]  f16  (33.7 MB)
  //   qkv_wh:  [1152,384]   f16  ( 0.9 MB)
  //   proj_wh: [384,384]    f16  ( 0.3 MB)
  //   q, k:    [B,H,352,32] f16  (34.6 MB each)
  //   vT:      [B,H,32,352] f16  (34.6 MB)
  //   ob:      [43904,384]  f16  (33.7 MB)
  //   biasc:   [12,343,343] f32  ( 5.6 MB)
  char* ws = (char*)d_ws;
  size_t off = 0;
  auto carve = [&](size_t bytes) -> void* {
    void* p = ws + off;
    off = (off + bytes + 255) & ~(size_t)255;
    return p;
  };
  const size_t xcnt  = (size_t)MROWS * CDIM;           // also ob element count
  const size_t qwcnt = (size_t)3 * CDIM * CDIM;
  const size_t pwcnt = (size_t)CDIM * CDIM;
  const size_t qsz   = (size_t)BATCH * HEADS * NPAD * HD * sizeof(half_t);
  const size_t bcsz  = (size_t)HEADS * WM_N * WM_N * sizeof(float);

  half_t* xh_t = (half_t*)carve(xcnt * sizeof(half_t));
  half_t* qw_t = (half_t*)carve(qwcnt * sizeof(half_t));
  half_t* pw_t = (half_t*)carve(pwcnt * sizeof(half_t));
  half_t* q_t  = (half_t*)carve(qsz);
  half_t* k_t  = (half_t*)carve(qsz);
  half_t* vT_t = (half_t*)carve(qsz);
  half_t* ob_t = (half_t*)carve(xcnt * sizeof(half_t));
  float*  bc_t = (float*)carve(bcsz);

  // one-time f32 -> f16 operand conversion (streaming, ~100 MB total)
  cvt16_kernel<<<(int)((xcnt / 4 + 255) / 256), 256, 0, stream>>>(x, xh_t, (int)xcnt);
  cvt16_kernel<<<(int)((qwcnt / 4 + 255) / 256), 256, 0, stream>>>(qkv_w, qw_t, (int)qwcnt);
  cvt16_kernel<<<(int)((pwcnt / 4 + 255) / 256), 256, 0, stream>>>(proj_w, pw_t, (int)pwcnt);

  const int NN = WM_N * WM_N;
  biasc_kernel<<<(NN + 255) / 256, 256, 0, stream>>>(rpb, rel, bc_t);
  zeropad_kernel<<<BATCH * HEADS, 288, 0, stream>>>(q_t, k_t, vT_t);
  qkv_kernel<<<dim3(MT32, 18), 32, 0, stream>>>(xh_t, qw_t, qkv_b, q_t, k_t, vT_t);
  attn_kernel<<<dim3(NPAD / 16, BATCH * HEADS), 32, 0, stream>>>(q_t, k_t, vT_t,
                                                                 bc_t, mask, ob_t);
  proj_kernel<<<dim3(MT32, 6), 32, 0, stream>>>(ob_t, pw_t, proj_b, out);
}
